// TAM_22067541967733
// MI455X (gfx1250) — matrix operand order repaired and checked by reading, not compiled, not run on hardware
//
#include <hip/hip_runtime.h>
#include <math.h>

typedef __bf16 bf16;
typedef __attribute__((ext_vector_type(16))) __bf16 v16bf;
typedef __attribute__((ext_vector_type(8)))  __bf16 v8bf;
typedef __attribute__((ext_vector_type(8)))  float  v8f;

// ---------------------------------------------------------------------------
// f32 -> bf16 conversion (weights, once per launch)
// ---------------------------------------------------------------------------
__global__ void f32_to_bf16_kernel(const float* __restrict__ src, bf16* __restrict__ dst, size_t n) {
    size_t i = (size_t)blockIdx.x * blockDim.x + threadIdx.x;
    if (i < n) dst[i] = (bf16)src[i];
}

// ---------------------------------------------------------------------------
// WMMA GEMM: C[M,N] = act( A[M,K](f32) @ W[N,K]^T(bf16) + bias[N] )
// One wave computes a 32(M) x 64(N) patch (2 M-tiles x 4 N-tiles): each
// B-fragment load feeds two v_wmma ops, halving L2-side W traffic vs 16x64.
// A is converted f32->bf16 in-register (v_cvt_pk_bf16_f32).
// M%32==0, N%64==0, K%32==0 required (all shapes here satisfy this).
// ---------------------------------------------------------------------------
__global__ void __launch_bounds__(256)
wmma_gemm(const float* __restrict__ A, const bf16* __restrict__ W,
          const float* __restrict__ bias, float* __restrict__ C,
          int M, int N, int K, int act)
{
    const int lane = threadIdx.x & 31;
    const int r    = lane & 15;
    const int half = lane >> 4;
    int wave = blockIdx.x * (blockDim.x >> 5) + (threadIdx.x >> 5);
    int n64  = N >> 6;
    int mT   = M >> 5;
    if (wave >= mT * n64) return;
    int m0 = (wave / n64) << 5;
    int n0 = (wave % n64) << 6;

    v8f acc[2][4];
    for (int j = 0; j < 4; ++j) {
        float bb = bias ? bias[n0 + j * 16 + r] : 0.f;
        for (int e = 0; e < 8; ++e) { acc[0][j][e] = bb; acc[1][j][e] = bb; }
    }

    const float* arow0 = A + (size_t)(m0 + r) * K + half * 16;
    const float* arow1 = arow0 + (size_t)16 * K;
    for (int k0 = 0; k0 < K; k0 += 32) {
        const float4* ap0 = (const float4*)(arow0 + k0);
        const float4* ap1 = (const float4*)(arow1 + k0);
        float4 x0 = ap0[0], x1 = ap0[1], x2 = ap0[2], x3 = ap0[3];
        float4 y0 = ap1[0], y1 = ap1[1], y2 = ap1[2], y3 = ap1[3];
        v16bf af0, af1;
        af0[0]=(bf16)x0.x; af0[1]=(bf16)x0.y; af0[2]=(bf16)x0.z; af0[3]=(bf16)x0.w;
        af0[4]=(bf16)x1.x; af0[5]=(bf16)x1.y; af0[6]=(bf16)x1.z; af0[7]=(bf16)x1.w;
        af0[8]=(bf16)x2.x; af0[9]=(bf16)x2.y; af0[10]=(bf16)x2.z; af0[11]=(bf16)x2.w;
        af0[12]=(bf16)x3.x; af0[13]=(bf16)x3.y; af0[14]=(bf16)x3.z; af0[15]=(bf16)x3.w;
        af1[0]=(bf16)y0.x; af1[1]=(bf16)y0.y; af1[2]=(bf16)y0.z; af1[3]=(bf16)y0.w;
        af1[4]=(bf16)y1.x; af1[5]=(bf16)y1.y; af1[6]=(bf16)y1.z; af1[7]=(bf16)y1.w;
        af1[8]=(bf16)y2.x; af1[9]=(bf16)y2.y; af1[10]=(bf16)y2.z; af1[11]=(bf16)y2.w;
        af1[12]=(bf16)y3.x; af1[13]=(bf16)y3.y; af1[14]=(bf16)y3.z; af1[15]=(bf16)y3.w;
        for (int j = 0; j < 4; ++j) {
            const v16bf bmat = *(const v16bf*)(W + (size_t)(n0 + j * 16 + r) * K + k0 + half * 16);
            acc[0][j] = __builtin_amdgcn_wmma_f32_16x16x32_bf16(
                false, af0, false, bmat, (short)0, acc[0][j], false, false);
            acc[1][j] = __builtin_amdgcn_wmma_f32_16x16x32_bf16(
                false, af1, false, bmat, (short)0, acc[1][j], false, false);
        }
    }

    for (int mi = 0; mi < 2; ++mi)
        for (int j = 0; j < 4; ++j)
            for (int e = 0; e < 8; ++e) {
                float v = acc[mi][j][e];
                if (act) v = tanhf(v);
                C[(size_t)(m0 + mi * 16 + half * 8 + e) * N + n0 + j * 16 + r] = v;
            }
}

// ---------------------------------------------------------------------------
// Script LSTM recurrence (one direction). Each block owns 16 batch rows.
// Per step: g = precomp_gates[n,t,:] + h_bf16 @ Whh^T  (WMMA, 8 waves x 8 tiles),
// then the gate nonlinearity + cell update runs fully in registers; new h is
// written as bf16 into an 8KB LDS tile for the next step's A-fragments.
// ---------------------------------------------------------------------------
__global__ void __launch_bounds__(256)
lstm_script_recur(const float* __restrict__ gates, const bf16* __restrict__ Whh,
                  float* __restrict__ outs, int T, int reverse, int dirOff)
{
    __shared__ __align__(32) bf16 hbf[16][256];
    const int tid  = threadIdx.x;
    const int lane = tid & 31;
    const int wv   = tid >> 5;
    const int r    = lane & 15;
    const int half = lane >> 4;
    const int j0   = wv * 32;          // hidden-col base owned by this wave
    const int n0   = blockIdx.x * 16;  // batch-row base for this block

    for (int i = tid; i < 16 * 256; i += 256) ((bf16*)hbf)[i] = (bf16)0.f;

    v8f cacc[2];
    for (int g = 0; g < 2; ++g)
        for (int e = 0; e < 8; ++e) cacc[g][e] = 0.f;

    for (int ts = 0; ts < T; ++ts) {
        int t = reverse ? (T - 1 - ts) : ts;
        __syncthreads();  // h tile (init or previous step) visible

        v8f gacc[4][2];
        for (int q = 0; q < 4; ++q)
            for (int g = 0; g < 2; ++g)
                for (int e = 0; e < 8; ++e) {
                    int row = half * 8 + e;
                    gacc[q][g][e] = gates[(((size_t)(n0 + row)) * T + t) * 1024
                                          + q * 256 + j0 + g * 16 + r];
                }

        for (int k0 = 0; k0 < 256; k0 += 32) {
            v8bf lo = *(const v8bf*)&hbf[r][k0 + half * 16];
            v8bf hi = *(const v8bf*)&hbf[r][k0 + half * 16 + 8];
            v16bf af;
            for (int e = 0; e < 8; ++e) { af[e] = lo[e]; af[8 + e] = hi[e]; }
            for (int q = 0; q < 4; ++q)
                for (int g = 0; g < 2; ++g) {
                    const v16bf bmat = *(const v16bf*)(Whh
                        + (size_t)(q * 256 + j0 + g * 16 + r) * 256 + k0 + half * 16);
                    gacc[q][g] = __builtin_amdgcn_wmma_f32_16x16x32_bf16(
                        false, af, false, bmat, (short)0, gacc[q][g], false, false);
                }
        }
        __syncthreads();  // all waves finished reading hbf

        for (int g = 0; g < 2; ++g)
            for (int e = 0; e < 8; ++e) {
                float ig = 1.f / (1.f + expf(-gacc[0][g][e]));
                float fg = 1.f / (1.f + expf(-gacc[1][g][e]));
                float gg = tanhf(gacc[2][g][e]);
                float og = 1.f / (1.f + expf(-gacc[3][g][e]));
                float c  = fg * cacc[g][e] + ig * gg;
                cacc[g][e] = c;
                float h = og * tanhf(c);
                int row = half * 8 + e;
                int col = j0 + g * 16 + r;
                hbf[row][col] = (bf16)h;
                outs[((size_t)(n0 + row) * T + t) * 512 + dirOff + col] = h;
            }
    }
}

// ---------------------------------------------------------------------------
// Scene-rank LSTM recurrence (B=1, latency-bound, tiny): block 0 = forward,
// block 1 = backward. h/c/gates live in LDS; f32 weights used directly.
// ---------------------------------------------------------------------------
__global__ void __launch_bounds__(256)
lstm_sr_recur(const float* __restrict__ gatesF, const float* __restrict__ gatesB,
              const float* __restrict__ WhhF, const float* __restrict__ WhhB,
              float* __restrict__ se, int T)
{
    __shared__ float h[256], c[256], gbuf[1024];
    int tid = threadIdx.x;
    int dir = blockIdx.x;
    const float* gates = dir ? gatesB : gatesF;
    const float* Whh   = dir ? WhhB : WhhF;
    int off = dir ? 256 : 0;
    h[tid] = 0.f; c[tid] = 0.f;
    __syncthreads();
    for (int ts = 0; ts < T; ++ts) {
        int t = dir ? (T - 1 - ts) : ts;
        for (int n = tid; n < 1024; n += 256) {
            float s = gates[(size_t)t * 1024 + n];
            const float* wr = Whh + (size_t)n * 256;
            for (int k = 0; k < 256; ++k) s += h[k] * wr[k];
            gbuf[n] = s;
        }
        __syncthreads();
        float ig = 1.f / (1.f + expf(-gbuf[tid]));
        float fg = 1.f / (1.f + expf(-gbuf[256 + tid]));
        float gg = tanhf(gbuf[512 + tid]);
        float og = 1.f / (1.f + expf(-gbuf[768 + tid]));
        float cc = fg * c[tid] + ig * gg;
        c[tid] = cc;
        float hh = og * tanhf(cc);
        h[tid] = hh;
        se[(size_t)t * 512 + off + tid] = hh;
        __syncthreads();
    }
}

// ---------------------------------------------------------------------------
// Script attention pooling: e=tanh(outs.attn_w), mask, softmax over T,
// weighted sum -> emb[:, 0:512]. One block per scene.
// ---------------------------------------------------------------------------
__global__ void __launch_bounds__(256)
script_attn_pool(const float* __restrict__ outs, const float* __restrict__ attn_w,
                 const int* __restrict__ lens, float* __restrict__ emb, int T)
{
    int n = blockIdx.x, tid = threadIdx.x;
    __shared__ float red[256];
    __shared__ float ebuf[128];
    __shared__ float stat[1];
    int part = tid / T;    // 2 partial sums per t
    int t    = tid % T;
    const float* orow = outs + ((size_t)n * T + t) * 512;
    float s = 0;
    for (int d = part * 256; d < part * 256 + 256; ++d) s += orow[d] * attn_w[d];
    red[tid] = s;
    __syncthreads();
    if (part == 0) {
        float v = tanhf(red[t] + red[T + t]);
        ebuf[t] = (t < lens[n]) ? v : -1e9f;
    }
    __syncthreads();
    if (tid == 0) {
        float m = -1e30f;
        for (int i = 0; i < T; ++i) m = fmaxf(m, ebuf[i]);
        float ss = 0;
        for (int i = 0; i < T; ++i) { ebuf[i] = expf(ebuf[i] - m); ss += ebuf[i]; }
        stat[0] = ss;
    }
    __syncthreads();
    float inv = 1.f / stat[0];
    for (int d = tid; d < 512; d += 256) {
        float acc = 0;
        for (int t2 = 0; t2 < T; ++t2) acc += ebuf[t2] * outs[((size_t)n * T + t2) * 512 + d];
        emb[(size_t)n * 1536 + d] = acc * inv;
    }
}

// ---------------------------------------------------------------------------
// Audio/vision attention pooling: ae=tanh(act.eW + eb), softmax over L,
// weighted sum -> emb[:, embOff:embOff+512]. One block per scene.
// ---------------------------------------------------------------------------
__global__ void __launch_bounds__(256)
mod_attn_pool(const float* __restrict__ act, const float* __restrict__ eW,
              const float* __restrict__ ebp, float* __restrict__ emb,
              int L, int embOff)
{
    int n = blockIdx.x, tid = threadIdx.x;
    __shared__ float red[256];
    __shared__ float ebuf[64];
    __shared__ float stat[1];
    int parts = 256 / L;
    int part  = tid / L;
    int l     = tid % L;
    int dlen  = 512 / parts;
    const float* arow = act + ((size_t)n * L + l) * 512;
    float s = 0;
    for (int d = part * dlen; d < (part + 1) * dlen; ++d) s += arow[d] * eW[d];
    red[tid] = s;
    __syncthreads();
    if (part == 0) {
        float tot = 0;
        for (int p = 0; p < parts; ++p) tot += red[p * L + l];
        ebuf[l] = tanhf(tot + ebp[0]);
    }
    __syncthreads();
    if (tid == 0) {
        float m = -1e30f;
        for (int i = 0; i < L; ++i) m = fmaxf(m, ebuf[i]);
        float ss = 0;
        for (int i = 0; i < L; ++i) { ebuf[i] = expf(ebuf[i] - m); ss += ebuf[i]; }
        stat[0] = ss;
    }
    __syncthreads();
    float inv = 1.f / stat[0];
    for (int d = tid; d < 512; d += 256) {
        float acc = 0;
        for (int l2 = 0; l2 < L; ++l2) acc += ebuf[l2] * act[((size_t)n * L + l2) * 512 + d];
        emb[(size_t)n * 1536 + embOff + d] = acc * inv;
    }
}

// ---------------------------------------------------------------------------
// Prefix sums over time of se -> P[(N+1)][512]
// ---------------------------------------------------------------------------
__global__ void prefix_time(const float* __restrict__ se, float* __restrict__ P, int N)
{
    int d = blockIdx.x * blockDim.x + threadIdx.x;
    if (d >= 512) return;
    float run = 0.f;
    P[d] = 0.f;
    for (int i = 0; i < N; ++i) {
        run += se[(size_t)i * 512 + d];
        P[(size_t)(i + 1) * 512 + d] = run;
    }
}

// ---------------------------------------------------------------------------
// Windowed means + interactions: builds u[N][4610]. One block per row.
// ---------------------------------------------------------------------------
__global__ void __launch_bounds__(256)
window_interact(const float* __restrict__ se, const float* __restrict__ P,
                float* __restrict__ u, int N)
{
    int i = blockIdx.x, tid = threadIdx.x;
    const int wlen = 51;  // int(0.1 * 512)
    __shared__ float red[5 * 256];
    __shared__ float cosv[2];
    int ls = max(i - wlen, 0);
    float dl = (float)max(i - ls, 1);
    int re = min(i + wlen, N);
    float dr = (float)max(re - i, 1);
    bool lastRow = (i == N - 1);

    float ss = 0, ll = 0, rr = 0, sl = 0, sr = 0;
    for (int d = tid; d < 512; d += 256) {
        float s = se[(size_t)i * 512 + d];
        float l = (P[(size_t)i * 512 + d] - P[(size_t)ls * 512 + d]) / dl;
        float r = lastRow ? 0.f : (P[(size_t)re * 512 + d] - P[(size_t)i * 512 + d]) / dr;
        ss += s * s; ll += l * l; rr += r * r; sl += s * l; sr += s * r;
    }
    red[tid] = ss; red[256 + tid] = ll; red[512 + tid] = rr;
    red[768 + tid] = sl; red[1024 + tid] = sr;
    __syncthreads();
    for (int o = 128; o > 0; o >>= 1) {
        if (tid < o)
            for (int q = 0; q < 5; ++q) red[q * 256 + tid] += red[q * 256 + tid + o];
        __syncthreads();
    }
    if (tid == 0) {
        const float eps = 1e-8f;
        float ns = fmaxf(sqrtf(red[0]), eps);
        float nl = fmaxf(sqrtf(red[256]), eps);
        float nr = fmaxf(sqrtf(red[512]), eps);
        cosv[0] = red[768] / (ns * nl);
        cosv[1] = red[1024] / (ns * nr);
    }
    __syncthreads();
    float* urow = u + (size_t)i * 4610;
    for (int d = tid; d < 512; d += 256) {
        float s = se[(size_t)i * 512 + d];
        float l = (P[(size_t)i * 512 + d] - P[(size_t)ls * 512 + d]) / dl;
        float r = lastRow ? 0.f : (P[(size_t)re * 512 + d] - P[(size_t)i * 512 + d]) / dr;
        urow[d] = s;
        urow[512 + d] = s;  urow[1024 + d] = l;  urow[1536 + d] = s * l;  urow[2048 + d] = s - l;
        urow[2561 + d] = s; urow[3073 + d] = r;  urow[3585 + d] = s * r;  urow[4097 + d] = s - r;
    }
    if (tid == 0) { urow[2560] = cosv[0]; urow[4609] = cosv[1]; }
}

// ---------------------------------------------------------------------------
// logits[N][5] = u @ tp_W^T + tp_b
// ---------------------------------------------------------------------------
__global__ void __launch_bounds__(256)
tp_logits(const float* __restrict__ u, const float* __restrict__ W,
          const float* __restrict__ b, float* __restrict__ logits)
{
    int i = blockIdx.x, tid = threadIdx.x;
    __shared__ float red[256];
    const float* urow = u + (size_t)i * 4610;
    for (int c = 0; c < 5; ++c) {
        const float* wr = W + (size_t)c * 4610;
        float s = 0;
        for (int d = tid; d < 4610; d += 256) s += urow[d] * wr[d];
        red[tid] = s;
        __syncthreads();
        for (int o = 128; o > 0; o >>= 1) {
            if (tid < o) red[tid] += red[tid + o];
            __syncthreads();
        }
        if (tid == 0) logits[(size_t)i * 5 + c] = red[0] + b[c];
        __syncthreads();
    }
}

// ---------------------------------------------------------------------------
// out[c][n] = softmax over n of logits[n][c]   (5 blocks)
// ---------------------------------------------------------------------------
__global__ void __launch_bounds__(256)
out_softmax(const float* __restrict__ logits, float* __restrict__ out, int N)
{
    int c = blockIdx.x, tid = threadIdx.x;
    __shared__ float red[256];
    float m = -1e30f;
    for (int n = tid; n < N; n += 256) m = fmaxf(m, logits[(size_t)n * 5 + c]);
    red[tid] = m;
    __syncthreads();
    for (int o = 128; o > 0; o >>= 1) {
        if (tid < o) red[tid] = fmaxf(red[tid], red[tid + o]);
        __syncthreads();
    }
    float mx = red[0];
    __syncthreads();
    float s = 0;
    for (int n = tid; n < N; n += 256) s += expf(logits[(size_t)n * 5 + c] - mx);
    red[tid] = s;
    __syncthreads();
    for (int o = 128; o > 0; o >>= 1) {
        if (tid < o) red[tid] += red[tid + o];
        __syncthreads();
    }
    float inv = 1.f / red[0];
    for (int n = tid; n < N; n += 256)
        out[(size_t)c * N + n] = expf(logits[(size_t)n * 5 + c] - mx) * inv;
}

// ---------------------------------------------------------------------------
extern "C" void kernel_launch(void* const* d_in, const int* in_sizes, int n_in,
                              void* d_out, int out_size, void* d_ws, size_t ws_size,
                              hipStream_t stream)
{
    (void)in_sizes; (void)n_in; (void)out_size; (void)ws_size;
    const int N = 512, T = 128, A = 64, V = 32;
    const int IN = 768, AUD = 128, VIS = 1024;

    const float* script   = (const float*)d_in[0];
    const int*   lens     = (const int*)d_in[1];
    const float* audio    = (const float*)d_in[2];
    const float* vision   = (const float*)d_in[3];
    const float* sc_Wih_f = (const float*)d_in[4];
    const float* sc_Whh_f = (const float*)d_in[5];
    const float* sc_b_f   = (const float*)d_in[6];
    const float* sc_Wih_b = (const float*)d_in[7];
    const float* sc_Whh_b = (const float*)d_in[8];
    const float* sc_b_b   = (const float*)d_in[9];
    const float* attn_w   = (const float*)d_in[10];
    const float* sr_Wih_f = (const float*)d_in[11];
    const float* sr_Whh_f = (const float*)d_in[12];
    const float* sr_b_f   = (const float*)d_in[13];
    const float* sr_Wih_b = (const float*)d_in[14];
    const float* sr_Whh_b = (const float*)d_in[15];
    const float* sr_b_b   = (const float*)d_in[16];
    const float* aud_W    = (const float*)d_in[17];
    const float* aud_b    = (const float*)d_in[18];
    const float* vis_W    = (const float*)d_in[19];
    const float* vis_b    = (const float*)d_in[20];
    const float* ea_W     = (const float*)d_in[21];
    const float* ea_b     = (const float*)d_in[22];
    const float* ev_W     = (const float*)d_in[23];
    const float* ev_b     = (const float*)d_in[24];
    const float* tp_W     = (const float*)d_in[25];
    const float* tp_b     = (const float*)d_in[26];

    char* ws = (char*)d_ws;
    size_t off = 0;
    auto alloc = [&](size_t bytes) -> void* {
        void* p = ws + off;
        off += (bytes + 255) & ~(size_t)255;
        return p;
    };

    float* gates_sc  = (float*)alloc((size_t)N * T * 1024 * 4);  // reused fwd then bwd
    float* outs      = (float*)alloc((size_t)N * T * 512 * 4);
    float* a_act     = (float*)alloc((size_t)N * A * 512 * 4);
    float* v_act     = (float*)alloc((size_t)N * V * 512 * 4);
    float* emb       = (float*)alloc((size_t)N * 1536 * 4);
    float* gates_srF = (float*)alloc((size_t)N * 1024 * 4);
    float* gates_srB = (float*)alloc((size_t)N * 1024 * 4);
    float* se        = (float*)alloc((size_t)N * 512 * 4);
    float* P         = (float*)alloc((size_t)(N + 1) * 512 * 4);
    float* u         = (float*)alloc((size_t)N * 4610 * 4);
    float* logits    = (float*)alloc((size_t)N * 5 * 4);
    bf16* scWihF_bf  = (bf16*)alloc((size_t)1024 * 768 * 2);
    bf16* scWihB_bf  = (bf16*)alloc((size_t)1024 * 768 * 2);
    bf16* scWhhF_bf  = (bf16*)alloc((size_t)1024 * 256 * 2);
    bf16* scWhhB_bf  = (bf16*)alloc((size_t)1024 * 256 * 2);
    bf16* audW_bf    = (bf16*)alloc((size_t)512 * 128 * 2);
    bf16* visW_bf    = (bf16*)alloc((size_t)512 * 1024 * 2);
    bf16* srWihF_bf  = (bf16*)alloc((size_t)1024 * 1536 * 2);
    bf16* srWihB_bf  = (bf16*)alloc((size_t)1024 * 1536 * 2);

    auto cvt = [&](const float* s, bf16* d, size_t n) {
        f32_to_bf16_kernel<<<(unsigned)((n + 255) / 256), 256, 0, stream>>>(s, d, n);
    };
    cvt(sc_Wih_f, scWihF_bf, (size_t)1024 * 768);
    cvt(sc_Wih_b, scWihB_bf, (size_t)1024 * 768);
    cvt(sc_Whh_f, scWhhF_bf, (size_t)1024 * 256);
    cvt(sc_Whh_b, scWhhB_bf, (size_t)1024 * 256);
    cvt(aud_W,    audW_bf,   (size_t)512 * 128);
    cvt(vis_W,    visW_bf,   (size_t)512 * 1024);
    cvt(sr_Wih_f, srWihF_bf, (size_t)1024 * 1536);
    cvt(sr_Wih_b, srWihB_bf, (size_t)1024 * 1536);

    auto gemm = [&](const float* Ap, const bf16* Wp, const float* bp, float* Cp,
                    int M_, int N_, int K_, int act) {
        int waves = (M_ / 32) * (N_ / 64);
        wmma_gemm<<<waves / 8, 256, 0, stream>>>(Ap, Wp, bp, Cp, M_, N_, K_, act);
    };

    // Script BiLSTM (gate buffer reused between directions; stream serializes)
    gemm(script, scWihF_bf, sc_b_f, gates_sc, N * T, 1024, IN, 0);
    lstm_script_recur<<<N / 16, 256, 0, stream>>>(gates_sc, scWhhF_bf, outs, T, 0, 0);
    gemm(script, scWihB_bf, sc_b_b, gates_sc, N * T, 1024, IN, 0);
    lstm_script_recur<<<N / 16, 256, 0, stream>>>(gates_sc, scWhhB_bf, outs, T, 1, 256);

    // Audio / vision projections (+tanh)
    gemm(audio,  audW_bf, aud_b, a_act, N * A, 512, AUD, 1);
    gemm(vision, visW_bf, vis_b, v_act, N * V, 512, VIS, 1);

    // Attention poolings -> emb[N][1536]
    script_attn_pool<<<N, 256, 0, stream>>>(outs, attn_w, lens, emb, T);
    mod_attn_pool<<<N, 256, 0, stream>>>(a_act, ea_W, ea_b, emb, A, 512);
    mod_attn_pool<<<N, 256, 0, stream>>>(v_act, ev_W, ev_b, emb, V, 1024);

    // Scene-rank BiLSTM
    gemm(emb, srWihF_bf, sr_b_f, gates_srF, N, 1024, 1536, 0);
    gemm(emb, srWihB_bf, sr_b_b, gates_srB, N, 1024, 1536, 0);
    lstm_sr_recur<<<2, 256, 0, stream>>>(gates_srF, gates_srB, sr_Whh_f, sr_Whh_b, se, N);

    // Windowed means, interactions, head, final softmax over scenes
    prefix_time<<<2, 256, 0, stream>>>(se, P, N);
    window_interact<<<N, 256, 0, stream>>>(se, P, u, N);
    tp_logits<<<N, 256, 0, stream>>>(u, tp_W, tp_b, logits);
    out_softmax<<<5, 256, 0, stream>>>(logits, (float*)d_out, N);
}
